// GraphAttentionLayer_47132971107178
// MI455X (gfx1250) — matrix operand order repaired and checked by reading, not compiled, not run on hardware
//
#include <hip/hip_runtime.h>
#include <hip/hip_bf16.h>

typedef __attribute__((ext_vector_type(16))) __bf16 v16bf;
typedef __attribute__((ext_vector_type(8)))  __bf16 v8bf;
typedef __attribute__((ext_vector_type(8)))  float  v8f;

#define FEAT   256
#define HEADS  4
#define CH     64
#define HC     256
#define NEG_SLOPE 0.2f
#define LN_EPS    1e-5f

// ---------- helpers ----------
__device__ __forceinline__ __bf16 f2bf(float f) {
  unsigned u = __float_as_uint(f);
  u = (u + 0x7FFFu + ((u >> 16) & 1u)) >> 16;       // round-to-nearest-even
  unsigned short s = (unsigned short)u;
  return __builtin_bit_cast(__bf16, s);
}

// monotonic float -> uint key for atomic max over floats
__device__ __forceinline__ unsigned f2key(float f) {
  unsigned u = __float_as_uint(f);
  return (u & 0x80000000u) ? ~u : (u | 0x80000000u);
}
__device__ __forceinline__ float key2f(unsigned k) {
  return (k & 0x80000000u) ? __uint_as_float(k & 0x7FFFFFFFu)
                           : __uint_as_float(~k);
}

__device__ __forceinline__ void fatomic_add(float* p, float v) {
  __hip_atomic_fetch_add(p, v, __ATOMIC_RELAXED, __HIP_MEMORY_SCOPE_AGENT);
}

// ---------- 0) init: out = bias, segmax = -inf key, segsum = 0 ----------
__global__ void gat_init(float* __restrict__ out, const float* __restrict__ bias,
                         unsigned* __restrict__ segmax, float* __restrict__ segsum,
                         int n_nodes) {
  long idx = (long)blockIdx.x * blockDim.x + threadIdx.x;
  long tot = (long)n_nodes * HC;
  if (idx < tot) out[idx] = bias[idx & (HC - 1)];
  if (idx < (long)n_nodes * HEADS) { segmax[idx] = 0u; segsum[idx] = 0.f; }
}

// ---------- 1) h = x @ W via bf16 WMMA (f32 accumulate) ----------
// 8 waves / block; each wave: one 16-row M tile x full 256 output columns.
// B fragments are software-pipelined (prefetch nt+1 before consuming nt) so the
// DS round-trip hides under the previous v_wmma instead of s_wait_dscnt 0.
__global__ __launch_bounds__(256) void gat_gemm(const float* __restrict__ x,
                                                const float* __restrict__ W,
                                                float* __restrict__ hbuf,
                                                int n_nodes) {
  __shared__ __bf16 Ws[256][40];   // W^T slab: [n][k_local], +8 pad (bank-conflict free)
  const int lane = threadIdx.x & 31;
  const int wave = threadIdx.x >> 5;
  const int ntiles = (n_nodes + 15) >> 4;
  const int tile = blockIdx.x * 8 + wave;
  const bool active = tile < ntiles;
  const int hi = lane >> 4;        // half-wave select
  const int li = lane & 15;
  long m = (long)tile * 16 + li;
  if (m >= n_nodes) m = n_nodes - 1;   // clamp loads; stores are guarded

  v8f acc[16];
  #pragma unroll
  for (int i = 0; i < 16; ++i)
    #pragma unroll
    for (int j = 0; j < 8; ++j) acc[i][j] = 0.f;

  for (int k0 = 0; k0 < FEAT; k0 += 32) {
    __syncthreads();
    // cooperative transpose-load of a 32x256 slab of W into LDS as bf16
    for (int idx = threadIdx.x; idx < 32 * 256; idx += 256) {
      int kk = idx >> 8;           // 0..31
      int n  = idx & 255;
      Ws[n][kk] = f2bf(W[(k0 + kk) * HC + n]);
    }
    __syncthreads();

    if (active) {
      // A fragment: 16-bit A 16x32 layout — lane holds K = k0+hi*8..+7 and k0+16+hi*8..+7
      const float* xr = x + m * FEAT + k0 + hi * 8;
      float4 r0 = *reinterpret_cast<const float4*>(xr);
      float4 r1 = *reinterpret_cast<const float4*>(xr + 4);
      float4 r2 = *reinterpret_cast<const float4*>(xr + 16);
      float4 r3 = *reinterpret_cast<const float4*>(xr + 20);
      v16bf a;
      a[0]  = f2bf(r0.x); a[1]  = f2bf(r0.y); a[2]  = f2bf(r0.z); a[3]  = f2bf(r0.w);
      a[4]  = f2bf(r1.x); a[5]  = f2bf(r1.y); a[6]  = f2bf(r1.z); a[7]  = f2bf(r1.w);
      a[8]  = f2bf(r2.x); a[9]  = f2bf(r2.y); a[10] = f2bf(r2.z); a[11] = f2bf(r2.w);
      a[12] = f2bf(r3.x); a[13] = f2bf(r3.y); a[14] = f2bf(r3.z); a[15] = f2bf(r3.w);

      // B fragment base for this lane: col N = nt*16+li, K = k0+hi*8..+7 and k0+16+hi*8..+7
      const __bf16* wcol = &Ws[li][hi * 8];          // nt stride = 16*40 elems = 1280 B
      v8bf nlo = *reinterpret_cast<const v8bf*>(wcol);
      v8bf nhi = *reinterpret_cast<const v8bf*>(wcol + 16);
      #pragma unroll
      for (int nt = 0; nt < 16; ++nt) {
        v8bf clo = nlo;
        v8bf chi = nhi;
        if (nt < 15) {                               // prefetch next tile's fragment
          const __bf16* wn = wcol + (nt + 1) * (16 * 40);
          nlo = *reinterpret_cast<const v8bf*>(wn);
          nhi = *reinterpret_cast<const v8bf*>(wn + 16);
        }
        v16bf b;
        #pragma unroll
        for (int i = 0; i < 8; ++i) { b[i] = clo[i]; b[8 + i] = chi[i]; }
        acc[nt] = __builtin_amdgcn_wmma_f32_16x16x32_bf16(
            false, a, false, b, (short)0, acc[nt], false, false);
      }
    }
  }

  if (active) {
    #pragma unroll
    for (int nt = 0; nt < 16; ++nt) {
      #pragma unroll
      for (int r = 0; r < 8; ++r) {
        long row = (long)tile * 16 + hi * 8 + r;
        if (row < n_nodes) hbuf[row * HC + nt * 16 + li] = acc[nt][r];
      }
    }
  }
}

// ---------- 2) a_src/a_dst per-head dot products (one wave per node) ----------
__global__ __launch_bounds__(256) void gat_attn(const float* __restrict__ hbuf,
                                                const float* __restrict__ att_src,
                                                const float* __restrict__ att_dst,
                                                float* __restrict__ a_src,
                                                float* __restrict__ a_dst,
                                                int n_nodes) {
  const int lane = threadIdx.x & 31;
  const int wave = threadIdx.x >> 5;
  const long node = (long)blockIdx.x * 8 + wave;
  if (node >= n_nodes) return;
  const float* hr = hbuf + node * HC + lane * 8;
  float4 v0 = *reinterpret_cast<const float4*>(hr);
  float4 v1 = *reinterpret_cast<const float4*>(hr + 4);
  const int head  = lane >> 3;
  const int cbase = head * CH + (lane & 7) * 8;
  float4 s0 = *reinterpret_cast<const float4*>(att_src + cbase);
  float4 s1 = *reinterpret_cast<const float4*>(att_src + cbase + 4);
  float4 d0 = *reinterpret_cast<const float4*>(att_dst + cbase);
  float4 d1 = *reinterpret_cast<const float4*>(att_dst + cbase + 4);
  float ps = v0.x*s0.x + v0.y*s0.y + v0.z*s0.z + v0.w*s0.w
           + v1.x*s1.x + v1.y*s1.y + v1.z*s1.z + v1.w*s1.w;
  float pd = v0.x*d0.x + v0.y*d0.y + v0.z*d0.z + v0.w*d0.w
           + v1.x*d1.x + v1.y*d1.y + v1.z*d1.z + v1.w*d1.w;
  #pragma unroll
  for (int msk = 1; msk < 8; msk <<= 1) {
    ps += __shfl_xor(ps, msk, 32);
    pd += __shfl_xor(pd, msk, 32);
  }
  if ((lane & 7) == 0) {
    a_src[node * HEADS + head] = ps;
    a_dst[node * HEADS + head] = pd;
  }
}

// ---------- 3) per-edge logits + leaky relu + segment max ----------
__global__ void gat_logits(const int* __restrict__ ei, const float* __restrict__ a_src,
                           const float* __restrict__ a_dst, float* __restrict__ alpha,
                           unsigned* __restrict__ segmax, int n_edges, int n_nodes) {
  long e = (long)blockIdx.x * blockDim.x + threadIdx.x;
  long etot = (long)n_edges + n_nodes;
  if (e >= etot) return;
  int s = (e < n_edges) ? ei[e] : (int)(e - n_edges);
  int d = (e < n_edges) ? ei[(long)n_edges + e] : (int)(e - n_edges);
  float4 as = *reinterpret_cast<const float4*>(a_src + (long)s * HEADS);
  float4 ad = *reinterpret_cast<const float4*>(a_dst + (long)d * HEADS);
  float v[4] = { as.x + ad.x, as.y + ad.y, as.z + ad.z, as.w + ad.w };
  #pragma unroll
  for (int h = 0; h < 4; ++h) v[h] = (v[h] >= 0.f) ? v[h] : NEG_SLOPE * v[h];
  *reinterpret_cast<float4*>(alpha + e * HEADS) = make_float4(v[0], v[1], v[2], v[3]);
  #pragma unroll
  for (int h = 0; h < 4; ++h)
    atomicMax(&segmax[(long)d * HEADS + h], f2key(v[h]));
}

// ---------- 4) exp(alpha - max) + segment sum ----------
__global__ void gat_expsum(float* __restrict__ alpha, const unsigned* __restrict__ segmax,
                           float* __restrict__ segsum, const int* __restrict__ ei,
                           int n_edges, int n_nodes) {
  long e = (long)blockIdx.x * blockDim.x + threadIdx.x;
  long etot = (long)n_edges + n_nodes;
  if (e >= etot) return;
  int d = (e < n_edges) ? ei[(long)n_edges + e] : (int)(e - n_edges);
  float4 a = *reinterpret_cast<const float4*>(alpha + e * HEADS);
  uint4 k = *reinterpret_cast<const uint4*>(segmax + (long)d * HEADS);
  float e0 = __expf(a.x - key2f(k.x));
  float e1 = __expf(a.y - key2f(k.y));
  float e2 = __expf(a.z - key2f(k.z));
  float e3 = __expf(a.w - key2f(k.w));
  *reinterpret_cast<float4*>(alpha + e * HEADS) = make_float4(e0, e1, e2, e3);
  fatomic_add(&segsum[(long)d * HEADS + 0], e0);
  fatomic_add(&segsum[(long)d * HEADS + 1], e1);
  fatomic_add(&segsum[(long)d * HEADS + 2], e2);
  fatomic_add(&segsum[(long)d * HEADS + 3], e3);
}

// ---------- 5) weighted message scatter (one wave per edge, atomics into out) ----------
__global__ __launch_bounds__(256) void gat_scatter(const int* __restrict__ ei,
                                                   const float* __restrict__ hbuf,
                                                   const float* __restrict__ alpha,
                                                   const float* __restrict__ segsum,
                                                   float* __restrict__ out,
                                                   int n_edges, int n_nodes) {
  const int lane = threadIdx.x & 31;
  const int wave = threadIdx.x >> 5;
  const long e = (long)blockIdx.x * 8 + wave;
  long etot = (long)n_edges + n_nodes;
  if (e >= etot) return;
  int s = (e < n_edges) ? ei[e] : (int)(e - n_edges);
  int d = (e < n_edges) ? ei[(long)n_edges + e] : (int)(e - n_edges);
  const int head = lane >> 3;
  float w = alpha[e * HEADS + head] /
            (segsum[(long)d * HEADS + head] + 1e-16f);
  const float* hr = hbuf + (long)s * HC + lane * 8;
  float4 m0 = *reinterpret_cast<const float4*>(hr);
  float4 m1 = *reinterpret_cast<const float4*>(hr + 4);
  float* orow = out + (long)d * HC + lane * 8;
  fatomic_add(orow + 0, w * m0.x);
  fatomic_add(orow + 1, w * m0.y);
  fatomic_add(orow + 2, w * m0.z);
  fatomic_add(orow + 3, w * m0.w);
  fatomic_add(orow + 4, w * m1.x);
  fatomic_add(orow + 5, w * m1.y);
  fatomic_add(orow + 6, w * m1.z);
  fatomic_add(orow + 7, w * m1.w);
}

// ---------- 6) LayerNorm in place (one wave per node) ----------
__global__ __launch_bounds__(256) void gat_ln(float* __restrict__ out,
                                              const float* __restrict__ gamma,
                                              const float* __restrict__ beta,
                                              int n_nodes) {
  const int lane = threadIdx.x & 31;
  const int wave = threadIdx.x >> 5;
  const long node = (long)blockIdx.x * 8 + wave;
  if (node >= n_nodes) return;
  float* row = out + node * HC + lane * 8;
  float4 v0 = *reinterpret_cast<const float4*>(row);
  float4 v1 = *reinterpret_cast<const float4*>(row + 4);
  float s  = v0.x + v0.y + v0.z + v0.w + v1.x + v1.y + v1.z + v1.w;
  float sq = v0.x*v0.x + v0.y*v0.y + v0.z*v0.z + v0.w*v0.w
           + v1.x*v1.x + v1.y*v1.y + v1.z*v1.z + v1.w*v1.w;
  #pragma unroll
  for (int msk = 1; msk < 32; msk <<= 1) {
    s  += __shfl_xor(s,  msk, 32);
    sq += __shfl_xor(sq, msk, 32);
  }
  float mean = s * (1.f / HC);
  float var  = sq * (1.f / HC) - mean * mean;
  float inv  = rsqrtf(var + LN_EPS);
  int c = lane * 8;
  float4 g0 = *reinterpret_cast<const float4*>(gamma + c);
  float4 g1 = *reinterpret_cast<const float4*>(gamma + c + 4);
  float4 b0 = *reinterpret_cast<const float4*>(beta + c);
  float4 b1 = *reinterpret_cast<const float4*>(beta + c + 4);
  float4 o0, o1;
  o0.x = (v0.x - mean) * inv * g0.x + b0.x;
  o0.y = (v0.y - mean) * inv * g0.y + b0.y;
  o0.z = (v0.z - mean) * inv * g0.z + b0.z;
  o0.w = (v0.w - mean) * inv * g0.w + b0.w;
  o1.x = (v1.x - mean) * inv * g1.x + b1.x;
  o1.y = (v1.y - mean) * inv * g1.y + b1.y;
  o1.z = (v1.z - mean) * inv * g1.z + b1.z;
  o1.w = (v1.w - mean) * inv * g1.w + b1.w;
  *reinterpret_cast<float4*>(row)     = o0;
  *reinterpret_cast<float4*>(row + 4) = o1;
}

// ---------- host launcher ----------
extern "C" void kernel_launch(void* const* d_in, const int* in_sizes, int n_in,
                              void* d_out, int out_size, void* d_ws, size_t ws_size,
                              hipStream_t stream) {
  const float* x       = (const float*)d_in[0];
  const int*   ei      = (const int*)d_in[1];
  const float* W       = (const float*)d_in[2];
  const float* att_src = (const float*)d_in[3];
  const float* att_dst = (const float*)d_in[4];
  const float* bias    = (const float*)d_in[5];
  const float* gamma   = (const float*)d_in[6];
  const float* beta    = (const float*)d_in[7];
  float* out = (float*)d_out;

  const int n_nodes = in_sizes[0] / FEAT;
  const int n_edges = in_sizes[1] / 2;
  const long etot = (long)n_edges + n_nodes;

  auto al = [](size_t v) { return (v + 255) & ~(size_t)255; };
  char* ws = (char*)d_ws;
  size_t o_h      = 0;
  size_t o_asrc   = al(o_h    + (size_t)n_nodes * HC * sizeof(float));
  size_t o_adst   = al(o_asrc + (size_t)n_nodes * HEADS * sizeof(float));
  size_t o_alpha  = al(o_adst + (size_t)n_nodes * HEADS * sizeof(float));
  size_t o_segmax = al(o_alpha + (size_t)etot * HEADS * sizeof(float));
  size_t o_segsum = al(o_segmax + (size_t)n_nodes * HEADS * sizeof(unsigned));

  float*    hbuf   = (float*)(ws + o_h);
  float*    a_src  = (float*)(ws + o_asrc);
  float*    a_dst  = (float*)(ws + o_adst);
  float*    alpha  = (float*)(ws + o_alpha);
  unsigned* segmax = (unsigned*)(ws + o_segmax);
  float*    segsum = (float*)(ws + o_segsum);

  {
    long tot = (long)n_nodes * HC;
    int blocks = (int)((tot + 255) / 256);
    gat_init<<<blocks, 256, 0, stream>>>(out, bias, segmax, segsum, n_nodes);
  }
  {
    int ntiles = (n_nodes + 15) / 16;
    gat_gemm<<<(ntiles + 7) / 8, 256, 0, stream>>>(x, W, hbuf, n_nodes);
  }
  gat_attn<<<(n_nodes + 7) / 8, 256, 0, stream>>>(hbuf, att_src, att_dst,
                                                  a_src, a_dst, n_nodes);
  gat_logits<<<(int)((etot + 255) / 256), 256, 0, stream>>>(ei, a_src, a_dst, alpha,
                                                            segmax, n_edges, n_nodes);
  gat_expsum<<<(int)((etot + 255) / 256), 256, 0, stream>>>(alpha, segmax, segsum, ei,
                                                            n_edges, n_nodes);
  gat_scatter<<<(int)((etot + 7) / 8), 256, 0, stream>>>(ei, hbuf, alpha, segsum, out,
                                                         n_edges, n_nodes);
  gat_ln<<<(n_nodes + 7) / 8, 256, 0, stream>>>(out, gamma, beta, n_nodes);
}
